// PokerTransformerNetwork_52158082843459
// MI455X (gfx1250) — compile-verified, matrix operand order checked by reading, not run on hardware
//
#include <hip/hip_runtime.h>
#include <hip/hip_bf16.h>

// ---------------------------------------------------------------------------
// Types for CDNA5 WMMA (wave32, v_wmma_f32_16x16x32_bf16)
// ---------------------------------------------------------------------------
typedef __bf16 bf16;
typedef __attribute__((ext_vector_type(16))) __bf16 v16bf;
typedef __attribute__((ext_vector_type(8)))  __bf16 v8bf;
typedef __attribute__((ext_vector_type(8)))  float  v8f;

#define N_EV 2048
#define DIM  256
#define NH   8
#define HD   2048   // NH*DIM
#define NL   6

__device__ __forceinline__ v8f wmma_bf16(v16bf a, v16bf b, v8f c) {
  // D = A(16x32) * B(32x16) + C, fp32 accumulate
  return __builtin_amdgcn_wmma_f32_16x16x32_bf16(false, a, false, b, (short)0, c,
                                                 false, false);
}

// Build a 16-element bf16 fragment from two contiguous 8-element (16B) chunks.
// Element e of an A/B fragment maps to K = (e>>3)*16 + g*8 + (e&7); the caller
// passes lo = &row[g*8], hi = &row[16 + g*8].
__device__ __forceinline__ v16bf mk_frag(const bf16* lo, const bf16* hi) {
  v8bf a = *(const v8bf*)lo;
  v8bf b = *(const v8bf*)hi;
  v16bf r;
#pragma unroll
  for (int i = 0; i < 8; ++i) { r[i] = a[i]; r[8 + i] = b[i]; }
  return r;
}

// ---------------------------------------------------------------------------
// Tensor Data Mover: async 2-D tile Global->LDS with built-in LDS row padding.
// D# built per CDNA5 ISA 8.3/8.4 (group0: count/lds/global/type=2,
// group1: data_size=2B, dims, tile dims, dim0 stride, pad fields).
// ---------------------------------------------------------------------------
#if defined(__HIP_DEVICE_COMPILE__) && __has_builtin(__builtin_amdgcn_tensor_load_to_lds)
#define HAVE_TDM 1
#else
#define HAVE_TDM 0
#endif

#if HAVE_TDM
typedef unsigned int u32x4 __attribute__((ext_vector_type(4)));
typedef int          i32x4 __attribute__((ext_vector_type(4)));
typedef int          i32x8 __attribute__((ext_vector_type(8)));

__device__ __forceinline__ unsigned lds_off(const void* p) {
  // low 32 bits of a flat LDS address are the LDS byte offset (ISA 10.2)
  return (unsigned)(unsigned long long)p;
}

// Load tile_h x tile_w bf16 elements (row stride = stride_elems) into LDS.
// pad_ic: pad every 2^(pad_ic+1) DWORDs; pad_ac: insert (pad_ac+1) DWORDs.
__device__ __forceinline__ void tdm_load_2d(unsigned lds_addr, const void* gptr,
                                            unsigned tile_w, unsigned tile_h,
                                            unsigned stride_elems,
                                            unsigned pad_ic, unsigned pad_ac) {
  unsigned long long ga = (unsigned long long)gptr;
  u32x4 g0;
  g0[0] = 1u;                                              // count=1, user D#
  g0[1] = lds_addr;                                        // LDS byte address
  g0[2] = (unsigned)ga;                                    // global addr lo32
  g0[3] = (unsigned)((ga >> 32) & 0x1FFFFFFu) | (2u << 30); // addr hi, type=2
  i32x8 g1;
  g1[0] = (int)((1u << 16) |                               // data_size = 2B
                (1u << 20) |                               // pad_enable
                (pad_ic << 22) | (pad_ac << 25));
  g1[1] = (int)((tile_w & 0xFFFFu) << 16);                 // tensor_dim0 lo16
  g1[2] = (int)(((tile_w >> 16) & 0xFFFFu) |               // tensor_dim0 hi16
                ((tile_h & 0xFFFFu) << 16));               // tensor_dim1 lo16
  g1[3] = (int)(((tile_h >> 16) & 0xFFFFu) |               // tensor_dim1 hi16
                ((tile_w & 0xFFFFu) << 16));               // tile_dim0
  g1[4] = (int)(tile_h & 0xFFFFu);                         // tile_dim1 (dim2=0)
  g1[5] = (int)stride_elems;                               // dim0 stride lo32
  g1[6] = 0;                                               // stride hi, d1s lo
  g1[7] = 0;
  i32x4 gz = {0, 0, 0, 0};
#if __clang_major__ >= 23
  i32x8 gz8 = {0, 0, 0, 0, 0, 0, 0, 0};
  __builtin_amdgcn_tensor_load_to_lds(g0, g1, gz, gz, gz8, 0);
#else
  __builtin_amdgcn_tensor_load_to_lds(g0, g1, gz, gz, 0);
#endif
}
#endif  // HAVE_TDM

// ---------------------------------------------------------------------------
// Elementwise helpers
// ---------------------------------------------------------------------------
__global__ void k_cvt_bf16(const float* __restrict__ s, bf16* __restrict__ d, int n) {
  int i = blockIdx.x * 256 + threadIdx.x;
  if (i < n) d[i] = (bf16)s[i];
}

// Fold Wh[2048,2048] over the H tiled copies of x: Wfold[j,d] = sum_h Wh[j,h*256+d]
__global__ void k_fold_wh(const float* __restrict__ Wh, bf16* __restrict__ Wf) {
  int i = blockIdx.x * 256 + threadIdx.x;     // i < 2048*256
  int j = i >> 8, d = i & 255;
  float s = 0.f;
#pragma unroll
  for (int h = 0; h < NH; ++h) s += Wh[(size_t)j * HD + h * DIM + d];
  Wf[(size_t)j * DIM + d] = (bf16)s;
}

// ---------------------------------------------------------------------------
// WMMA GEMM:  C[M,N] = act(A[M,K] @ W[N,K]^T + bias[N])
// A,W bf16 row-major. Block 256 thr (8 waves), tile 128x128, wave tile 32x64,
// K staged by 32 via double-buffered TDM DMA (fallback: manual b128 staging).
// ---------------------------------------------------------------------------
#define LDSP 40   // 32 + 8 bf16 padding (80B rows, 16B aligned)

__global__ __launch_bounds__(256) void k_gemm(
    const bf16* __restrict__ A, const bf16* __restrict__ W,
    const float* __restrict__ bias, float* __restrict__ Cf, bf16* __restrict__ Cb,
    int M, int N, int K, int relu) {
  __shared__ bf16 sA[2][128 * LDSP];
  __shared__ bf16 sB[2][128 * LDSP];
  const int t  = threadIdx.x;
  const int l  = t & 31;
  const int g  = (l >> 4) & 1;
  const int ml = l & 15;
  const int wid = t >> 5, wm = wid & 3, wn = wid >> 2;   // 4x2 wave grid
  const int bm = blockIdx.x * 128, bn = blockIdx.y * 128;

  v8f acc[2][4];
#pragma unroll
  for (int mi = 0; mi < 2; ++mi)
#pragma unroll
    for (int c = 0; c < 4; ++c) acc[mi][c] = (v8f){0, 0, 0, 0, 0, 0, 0, 0};

  const int nk = K >> 5;

#if HAVE_TDM
  if (wid == 0) {
    // pad every 16 DWORDs (64B row) by 4 DWORDs -> LDSP=40 layout
    tdm_load_2d(lds_off(&sA[0][0]), A + (size_t)bm * K, 32, 128, K, 3, 3);
    tdm_load_2d(lds_off(&sB[0][0]), W + (size_t)bn * K, 32, 128, K, 3, 3);
  }
#endif

  for (int kt = 0; kt < nk; ++kt) {
    const int cur = kt & 1;
#if HAVE_TDM
    if (wid == 0) {
      if (kt + 1 < nk) {
        tdm_load_2d(lds_off(&sA[cur ^ 1][0]),
                    A + (size_t)bm * K + (kt + 1) * 32, 32, 128, K, 3, 3);
        tdm_load_2d(lds_off(&sB[cur ^ 1][0]),
                    W + (size_t)bn * K + (kt + 1) * 32, 32, 128, K, 3, 3);
        __builtin_amdgcn_s_wait_tensorcnt(2);   // oldest pair (cur) complete
      } else {
        __builtin_amdgcn_s_wait_tensorcnt(0);
      }
    }
    __syncthreads();
#else
    {
      const int row = t >> 2, ch = t & 3;
#pragma unroll
      for (int i = 0; i < 2; ++i) {
        int rw = row + 64 * i;
        *(v8bf*)&sA[cur][rw * LDSP + ch * 8] =
            *(const v8bf*)(A + (size_t)(bm + rw) * K + kt * 32 + ch * 8);
        *(v8bf*)&sB[cur][rw * LDSP + ch * 8] =
            *(const v8bf*)(W + (size_t)(bn + rw) * K + kt * 32 + ch * 8);
      }
      if (kt + 1 < nk) {
        __builtin_prefetch(A + (size_t)(bm + row) * K + kt * 32 + 32, 0, 1);
        __builtin_prefetch(W + (size_t)(bn + row) * K + kt * 32 + 32, 0, 1);
      }
    }
    __syncthreads();
#endif

    v16bf bfr[4];
#pragma unroll
    for (int c = 0; c < 4; ++c) {
      const bf16* br = &sB[cur][(wn * 64 + c * 16 + ml) * LDSP + g * 8];
      bfr[c] = mk_frag(br, br + 16);
    }
#pragma unroll
    for (int mi = 0; mi < 2; ++mi) {
      const bf16* ar = &sA[cur][(wm * 32 + mi * 16 + ml) * LDSP + g * 8];
      v16bf af = mk_frag(ar, ar + 16);
#pragma unroll
      for (int c = 0; c < 4; ++c)
        acc[mi][c] = wmma_bf16(af, bfr[c], acc[mi][c]);
    }
    __syncthreads();
  }

#pragma unroll
  for (int c = 0; c < 4; ++c) {
    int ncol = bn + wn * 64 + c * 16 + ml;
    float bv = bias[ncol];
#pragma unroll
    for (int mi = 0; mi < 2; ++mi)
#pragma unroll
      for (int r = 0; r < 8; ++r) {
        int mrow = bm + wm * 32 + mi * 16 + r + 8 * g;   // C layout: M = r + 8g
        float v = acc[mi][c][r] + bv;
        if (relu) v = fmaxf(v, 0.f);
        if (Cf) Cf[(size_t)mrow * N + ncol] = v;
        if (Cb) Cb[(size_t)mrow * N + ncol] = (bf16)v;
      }
  }
}

// ---------------------------------------------------------------------------
// Flash attention: Q=K=V=hs per head, causal. One wave per (head, 16-row tile).
// Computes S^T = K @ Q^T so the probability A-fragment for P@V is identity with
// the S^T C-layout. KV tiles staged by double-buffered TDM DMA.
// ---------------------------------------------------------------------------
#define KVS 264   // 256 + 8 pad

__global__ __launch_bounds__(32) void k_attn(const bf16* __restrict__ hs,
                                             float* __restrict__ O) {
  __shared__ bf16 kv[2][32 * KVS];
  const int qi = blockIdx.x;        // 0..127
  const int h  = blockIdx.y;        // 0..7
  const int q0 = qi * 16;
  const int l  = threadIdx.x;
  const int g  = l >> 4;
  const int ml = l & 15;

  // Q fragments (B-layout): lane (g,ml) holds Q row q0+ml over d
  v16bf qf[8];
  const bf16* qrow = hs + (size_t)(q0 + ml) * HD + h * DIM;
#pragma unroll
  for (int kc = 0; kc < 8; ++kc)
    qf[kc] = mk_frag(qrow + kc * 32 + g * 8, qrow + kc * 32 + 16 + g * 8);

  v8f o[16];
#pragma unroll
  for (int c = 0; c < 16; ++c) o[c] = (v8f){0, 0, 0, 0, 0, 0, 0, 0};
  float m_row = -1e30f, l_row = 0.f;

  const int jt = (q0 + 16 + 31) >> 5;          // 32-key tiles needed (causal)

#if HAVE_TDM
  // pad every 128 DWORDs (512B row) by 4 DWORDs -> KVS=264 layout
  tdm_load_2d(lds_off(&kv[0][0]), hs + h * DIM, 256, 32, HD, 6, 3);
#endif

  for (int j = 0; j < jt; ++j) {
    const int cur = j & 1;
#if HAVE_TDM
    __syncthreads();   // drain LDS reads of the buffer about to be overwritten
    if (j + 1 < jt) {
      tdm_load_2d(lds_off(&kv[cur ^ 1][0]),
                  hs + (size_t)((j + 1) * 32) * HD + h * DIM, 256, 32, HD, 6, 3);
      __builtin_amdgcn_s_wait_tensorcnt(1);    // oldest (cur) complete
    } else {
      __builtin_amdgcn_s_wait_tensorcnt(0);
    }
    __syncthreads();
#else
    // stage KV tile [32 keys x 256 d] (coalesced 16B per lane per row)
#pragma unroll 4
    for (int tt = 0; tt < 32; ++tt)
      *(v8bf*)&kv[cur][tt * KVS + l * 8] =
          *(const v8bf*)(hs + (size_t)(j * 32 + tt) * HD + h * DIM + l * 8);
    __syncthreads();
#endif

    // S^T tiles: 32 keys x 16 rows = two 16x16 C tiles
    float p[2][8];
    float mloc = -1e30f;
#pragma unroll
    for (int st = 0; st < 2; ++st) {
      v8f s = (v8f){0, 0, 0, 0, 0, 0, 0, 0};
#pragma unroll
      for (int kc = 0; kc < 8; ++kc) {
        const bf16* kr = &kv[cur][(st * 16 + ml) * KVS + kc * 32 + g * 8];
        v16bf kf = mk_frag(kr, kr + 16);       // A: K rows over d
        s = wmma_bf16(kf, qf[kc], s);          // B: Q^T columns = Q rows
      }
#pragma unroll
      for (int r = 0; r < 8; ++r) {
        float sv = s[r] * 0.0625f;             // 1/sqrt(256)
        int key = j * 32 + st * 16 + r + 8 * g;
        if (key > q0 + ml) sv = -1e30f;        // causal mask
        p[st][r] = sv;
        mloc = fmaxf(mloc, sv);
      }
    }
    mloc = fmaxf(mloc, __shfl_xor(mloc, 16));
    float mnew  = fmaxf(m_row, mloc);
    float alpha = __expf(m_row - mnew);
    m_row = mnew;

    float rsum = 0.f;
#pragma unroll
    for (int st = 0; st < 2; ++st)
#pragma unroll
      for (int r = 0; r < 8; ++r) {
        float e = __expf(p[st][r] - mnew);
        p[st][r] = e;
        rsum += e;
      }
    rsum += __shfl_xor(rsum, 16);
    l_row = l_row * alpha + rsum;

    // rescale O by alpha of each output row (rows r + 8g)
    float al[8];
#pragma unroll
    for (int r = 0; r < 8; ++r) al[r] = __shfl(alpha, 8 * g + r);
#pragma unroll
    for (int c = 0; c < 16; ++c)
#pragma unroll
      for (int r = 0; r < 8; ++r) o[c][r] *= al[r];

    // P fragment (A-layout) == S^T C-layout: purely in-lane
    v16bf pf;
#pragma unroll
    for (int e = 0; e < 16; ++e) pf[e] = (bf16)p[e >> 3][e & 7];

    // O += P @ V  (16 column chunks)
#pragma unroll
    for (int c = 0; c < 16; ++c) {
      v16bf vf;
#pragma unroll
      for (int e = 0; e < 16; ++e) {
        int kk = (e >> 3) * 16 + g * 8 + (e & 7);
        vf[e] = kv[cur][kk * KVS + c * 16 + ml];
      }
      o[c] = wmma_bf16(pf, vf, o[c]);
    }
  }

  float li[8];
#pragma unroll
  for (int r = 0; r < 8; ++r) li[r] = 1.f / __shfl(l_row, 8 * g + r);
#pragma unroll
  for (int c = 0; c < 16; ++c)
#pragma unroll
    for (int r = 0; r < 8; ++r)
      O[(size_t)(q0 + r + 8 * g) * HD + h * DIM + c * 16 + ml] = o[c][r] * li[r];
}

// ---------------------------------------------------------------------------
// Block reductions (256 threads = 8 waves)
// ---------------------------------------------------------------------------
__device__ __forceinline__ float block_red(float v, float* red, bool isMax) {
#pragma unroll
  for (int o = 16; o; o >>= 1) {
    float t = __shfl_xor(v, o);
    v = isMax ? fmaxf(v, t) : v + t;
  }
  int wid = threadIdx.x >> 5;
  if ((threadIdx.x & 31) == 0) red[wid] = v;
  __syncthreads();
  int nw = (blockDim.x + 31) >> 5;
  float r = isMax ? -3e38f : 0.f;
  for (int i = 0; i < nw; ++i) r = isMax ? fmaxf(r, red[i]) : r + red[i];
  __syncthreads();
  return r;
}

// LN(out[n,h,:] + x[n,:]) -> normed bf16 [N, H*D]
__global__ __launch_bounds__(256) void k_ln_attn(
    const float* __restrict__ O, const float* __restrict__ x,
    const float* __restrict__ gm, const float* __restrict__ bt,
    bf16* __restrict__ out) {
  __shared__ float red[8];
  int n = blockIdx.x, d = threadIdx.x;
  float xv = x[(size_t)n * DIM + d];
  float gd = gm[d], bd = bt[d];
  for (int h = 0; h < NH; ++h) {
    float v = O[(size_t)n * HD + h * DIM + d] + xv;
    float mean = block_red(v, red, false) * (1.f / DIM);
    float c = v - mean;
    float var = block_red(c * c, red, false) * (1.f / DIM);
    float y = c * rsqrtf(var + 1e-5f) * gd + bd;
    out[(size_t)n * HD + h * DIM + d] = (bf16)y;
  }
}

// LN(f2[n,:] + a[n,:]) -> x (f32 + bf16)
__global__ __launch_bounds__(256) void k_ln_ffn(
    const float* __restrict__ f2, const float* __restrict__ a,
    const float* __restrict__ gm, const float* __restrict__ bt,
    float* __restrict__ xo, bf16* __restrict__ xb) {
  __shared__ float red[8];
  int n = blockIdx.x, d = threadIdx.x;
  float v = f2[(size_t)n * DIM + d] + a[(size_t)n * DIM + d];
  float mean = block_red(v, red, false) * (1.f / DIM);
  float c = v - mean;
  float var = block_red(c * c, red, false) * (1.f / DIM);
  float y = c * rsqrtf(var + 1e-5f) * gm[d] + bt[d];
  xo[(size_t)n * DIM + d] = y;
  xb[(size_t)n * DIM + d] = (bf16)y;
}

// pcs = LinearBlock(current_state): 128 -> relu 512 -> 256
__global__ __launch_bounds__(512) void k_pcs(
    const float* __restrict__ cs, const float* __restrict__ W1,
    const float* __restrict__ b1, const float* __restrict__ W2,
    const float* __restrict__ b2, float* __restrict__ pcs) {
  __shared__ float css[128];
  __shared__ float hb[512];
  int t = threadIdx.x;
  if (t < 128) css[t] = cs[t];
  __syncthreads();
  float a = b1[t];
  for (int k = 0; k < 128; ++k) a += css[k] * W1[t * 128 + k];
  hb[t] = fmaxf(a, 0.f);
  __syncthreads();
  if (t < 256) {
    float o = b2[t];
    for (int j = 0; j < 512; ++j) o += hb[j] * W2[t * 512 + j];
    pcs[t] = o;
  }
}

// sv[n] = dot(x[n,:], pcs)/16
__global__ __launch_bounds__(256) void k_sv(const float* __restrict__ x,
                                            const float* __restrict__ pcs,
                                            float* __restrict__ sv) {
  __shared__ float red[8];
  int n = blockIdx.x, d = threadIdx.x;
  float s = block_red(x[(size_t)n * DIM + d] * pcs[d], red, false);
  if (d == 0) sv[n] = s * 0.0625f;
}

// softmax over sv[2048], fpb[d] = sum_n w[n] * x[n,d]
__global__ __launch_bounds__(256) void k_smax_fpb(
    const float* __restrict__ sv, const float* __restrict__ x,
    float* __restrict__ wbuf, float* __restrict__ fpb) {
  __shared__ float red[8];
  int t = threadIdx.x;
  float m = -1e30f;
  for (int n = t; n < N_EV; n += 256) m = fmaxf(m, sv[n]);
  m = block_red(m, red, true);
  float s = 0.f;
  for (int n = t; n < N_EV; n += 256) {
    float e = __expf(sv[n] - m);
    wbuf[n] = e;
    s += e;
  }
  s = block_red(s, red, false);
  float inv = 1.f / s;
  float acc = 0.f;
  for (int n = 0; n < N_EV; ++n) acc += wbuf[n] * x[(size_t)n * DIM + t];
  fpb[t] = acc * inv;
}

// out = LinearBlock(concat(fpb, pcs)): 512 -> relu 256 -> 256
__global__ __launch_bounds__(512) void k_final(
    const float* __restrict__ fpb, const float* __restrict__ pcs,
    const float* __restrict__ W1, const float* __restrict__ b1,
    const float* __restrict__ W2, const float* __restrict__ b2,
    float* __restrict__ out) {
  __shared__ float cat[512];
  __shared__ float hb[256];
  int t = threadIdx.x;
  cat[t] = (t < 256) ? fpb[t] : pcs[t - 256];
  __syncthreads();
  if (t < 256) {
    float a = b1[t];
    for (int k = 0; k < 512; ++k) a += cat[k] * W1[t * 512 + k];
    hb[t] = fmaxf(a, 0.f);
  }
  __syncthreads();
  if (t < 256) {
    float o = b2[t];
    for (int j = 0; j < 256; ++j) o += hb[j] * W2[t * 256 + j];
    out[t] = o;
  }
}

// ---------------------------------------------------------------------------
// Host orchestration
// ---------------------------------------------------------------------------
extern "C" void kernel_launch(void* const* d_in, const int* in_sizes, int n_in,
                              void* d_out, int out_size, void* d_ws, size_t ws_size,
                              hipStream_t stream) {
  (void)in_sizes; (void)n_in; (void)out_size; (void)ws_size;
  const float* previous_bets = (const float*)d_in[0];
  const float* current_state = (const float*)d_in[1];
  const float* pb_W1 = (const float*)d_in[2];
  const float* pb_b1 = (const float*)d_in[3];
  const float* pb_W2 = (const float*)d_in[4];
  const float* pb_b2 = (const float*)d_in[5];
  const float* cs_W1 = (const float*)d_in[6];
  const float* cs_b1 = (const float*)d_in[7];
  const float* cs_W2 = (const float*)d_in[8];
  const float* cs_b2 = (const float*)d_in[9];
  const float* blk_Wh = (const float*)d_in[10];
  const float* blk_bh = (const float*)d_in[11];
  const float* blk_Wr = (const float*)d_in[12];
  const float* blk_br = (const float*)d_in[13];
  const float* blk_Wf1 = (const float*)d_in[14];
  const float* blk_bf1 = (const float*)d_in[15];
  const float* blk_Wf2 = (const float*)d_in[16];
  const float* blk_bf2 = (const float*)d_in[17];
  const float* blk_g = (const float*)d_in[18];
  const float* blk_b = (const float*)d_in[19];
  const float* fin_W1 = (const float*)d_in[20];
  const float* fin_b1 = (const float*)d_in[21];
  const float* fin_W2 = (const float*)d_in[22];
  const float* fin_b2 = (const float*)d_in[23];

  char* wsb = (char*)d_ws;
  size_t off = 0;
  auto alloc = [&](size_t bytes) -> void* {
    void* p = wsb + off;
    off = (off + bytes + 255) & ~(size_t)255;
    return p;
  };
  float* x_f   = (float*)alloc((size_t)N_EV * DIM * 4);
  bf16*  x_b   = (bf16*) alloc((size_t)N_EV * DIM * 2);
  bf16*  hs_b  = (bf16*) alloc((size_t)N_EV * HD * 2);
  float* O_f   = (float*)alloc((size_t)N_EV * HD * 4);
  bf16*  nrm_b = (bf16*) alloc((size_t)N_EV * HD * 2);
  float* a_f   = (float*)alloc((size_t)N_EV * DIM * 4);
  bf16*  a_b   = (bf16*) alloc((size_t)N_EV * DIM * 2);
  bf16*  f1_b  = (bf16*) alloc((size_t)N_EV * DIM * 2);
  float* f2_f  = (float*)alloc((size_t)N_EV * DIM * 4);
  bf16*  h1_b  = (bf16*) alloc((size_t)N_EV * 512 * 2);
  bf16*  pb_b  = (bf16*) alloc((size_t)N_EV * 128 * 2);
  bf16*  wpb1  = (bf16*) alloc((size_t)512 * 128 * 2);
  bf16*  wpb2  = (bf16*) alloc((size_t)256 * 512 * 2);
  bf16*  wfold = (bf16*) alloc((size_t)NL * HD * DIM * 2);
  bf16*  wr_b  = (bf16*) alloc((size_t)NL * DIM * HD * 2);
  bf16*  wf1_b = (bf16*) alloc((size_t)NL * DIM * DIM * 2);
  bf16*  wf2_b = (bf16*) alloc((size_t)NL * DIM * DIM * 2);
  float* pcs   = (float*)alloc(256 * 4);
  float* sv    = (float*)alloc(N_EV * 4);
  float* wbuf  = (float*)alloc(N_EV * 4);
  float* fpb   = (float*)alloc(256 * 4);

  auto cvt = [&](const float* s, bf16* d, int n) {
    k_cvt_bf16<<<(n + 255) / 256, 256, 0, stream>>>(s, d, n);
  };

  // weight/activation conversions + Wh folding (cheap vs. 23.3 TB/s HBM)
  cvt(previous_bets, pb_b, N_EV * 128);
  cvt(pb_W1, wpb1, 512 * 128);
  cvt(pb_W2, wpb2, 256 * 512);
  for (int l = 0; l < NL; ++l) {
    k_fold_wh<<<HD * DIM / 256, 256, 0, stream>>>(blk_Wh + (size_t)l * HD * HD,
                                                  wfold + (size_t)l * HD * DIM);
    cvt(blk_Wr + (size_t)l * DIM * HD, wr_b + (size_t)l * DIM * HD, DIM * HD);
    cvt(blk_Wf1 + (size_t)l * DIM * DIM, wf1_b + (size_t)l * DIM * DIM, DIM * DIM);
    cvt(blk_Wf2 + (size_t)l * DIM * DIM, wf2_b + (size_t)l * DIM * DIM, DIM * DIM);
  }

  // input projection: x = relu(pb @ W1^T + b1) @ W2^T + b2
  k_gemm<<<dim3(N_EV / 128, 512 / 128), 256, 0, stream>>>(
      pb_b, wpb1, pb_b1, nullptr, h1_b, N_EV, 512, 128, 1);
  k_gemm<<<dim3(N_EV / 128, DIM / 128), 256, 0, stream>>>(
      h1_b, wpb2, pb_b2, x_f, x_b, N_EV, DIM, 512, 0);

  for (int l = 0; l < NL; ++l) {
    const bf16* Wf = wfold + (size_t)l * HD * DIM;
    // hs = x @ Wfold^T + bh   (folded tile(x,H) @ Wh^T)
    k_gemm<<<dim3(N_EV / 128, HD / 128), 256, 0, stream>>>(
        x_b, Wf, blk_bh + (size_t)l * HD, nullptr, hs_b, N_EV, HD, DIM, 0);
    // causal multi-head self-attention (flash, TDM-staged KV)
    k_attn<<<dim3(N_EV / 16, NH), 32, 0, stream>>>(hs_b, O_f);
    // LN(attn_out + x)
    k_ln_attn<<<N_EV, 256, 0, stream>>>(O_f, x_f, blk_g + l * DIM,
                                        blk_b + l * DIM, nrm_b);
    // a = normed @ Wr^T + br
    k_gemm<<<dim3(N_EV / 128, DIM / 128), 256, 0, stream>>>(
        nrm_b, wr_b + (size_t)l * DIM * HD, blk_br + l * DIM, a_f, a_b,
        N_EV, DIM, HD, 0);
    // FFN
    k_gemm<<<dim3(N_EV / 128, DIM / 128), 256, 0, stream>>>(
        a_b, wf1_b + (size_t)l * DIM * DIM, blk_bf1 + l * DIM, nullptr, f1_b,
        N_EV, DIM, DIM, 1);
    k_gemm<<<dim3(N_EV / 128, DIM / 128), 256, 0, stream>>>(
        f1_b, wf2_b + (size_t)l * DIM * DIM, blk_bf2 + l * DIM, f2_f, nullptr,
        N_EV, DIM, DIM, 0);
    // x = LN(f + a)
    k_ln_ffn<<<N_EV, 256, 0, stream>>>(f2_f, a_f, blk_g + l * DIM,
                                       blk_b + l * DIM, x_f, x_b);
  }

  // tail: pcs, softmax pooling, final head
  k_pcs<<<1, 512, 0, stream>>>(current_state, cs_W1, cs_b1, cs_W2, cs_b2, pcs);
  k_sv<<<N_EV, 256, 0, stream>>>(x_f, pcs, sv);
  k_smax_fpb<<<1, 256, 0, stream>>>(sv, x_f, wbuf, fpb);
  k_final<<<1, 512, 0, stream>>>(fpb, pcs, fin_W1, fin_b1, fin_W2, fin_b2,
                                 (float*)d_out);
}
